// DecoderModel_11063835754609
// MI455X (gfx1250) — matrix operand order repaired and checked by reading, not compiled, not run on hardware
//
#include <hip/hip_runtime.h>
#include <stdint.h>

typedef __bf16 bf16t;
typedef __bf16 v16bf __attribute__((ext_vector_type(16)));
typedef __bf16 v8bf  __attribute__((ext_vector_type(8)));
typedef float  v8f   __attribute__((ext_vector_type(8)));

#define B_ 2
#define S_ 2048
#define D_ 1024
#define V_ 32000

// ---------- helpers ----------
__device__ __forceinline__ bf16t f2bf(float f) {
  union { float f; uint32_t u; } in; in.f = f;
  uint32_t u = in.u;
  uint32_t r = (u + 0x7FFFu + ((u >> 16) & 1u)) >> 16;   // RNE f32->bf16
  union { uint16_t us; bf16t b; } out; out.us = (uint16_t)r;
  return out.b;
}

// ---------- f32 -> bf16 conversion ----------
__global__ __launch_bounds__(256) void cvt_bf16_kernel(const float* __restrict__ in,
                                                       bf16t* __restrict__ out, long n) {
  long i = (long)blockIdx.x * blockDim.x + threadIdx.x;
  if (i < n) out[i] = f2bf(in[i]);
}

// ---------- embedding + positional ----------
__global__ __launch_bounds__(256) void embed_kernel(const int* __restrict__ x,
                                                    const float* __restrict__ emb,
                                                    const float* __restrict__ pos,
                                                    float* __restrict__ outf,
                                                    bf16t* __restrict__ outb,
                                                    long total) {
  long i = (long)blockIdx.x * blockDim.x + threadIdx.x;
  if (i >= total) return;
  int d = (int)(i % D_);
  long t = i / D_;            // b*S + s
  int s = (int)(t % S_);
  int tok = x[t];
  float v = emb[(long)tok * D_ + d] + pos[(long)s * D_ + d];
  outf[i] = v;
  outb[i] = f2bf(v);
}

// ---------- fused add + LayerNorm (writes f32 residual + bf16 activation) ----------
__global__ __launch_bounds__(256) void add_ln_kernel(const float* __restrict__ x,
                                                     const float* __restrict__ y,
                                                     const float* __restrict__ g,
                                                     const float* __restrict__ bta,
                                                     float* __restrict__ outf,
                                                     bf16t* __restrict__ outb) {
  long row = blockIdx.x;
  const float* xr = x + row * D_;
  const float* yr = y + row * D_;
  __shared__ float red[256];
  float s = 0.f, s2 = 0.f;
  for (int d = threadIdx.x; d < D_; d += 256) {
    float v = xr[d] + yr[d];
    s += v; s2 += v * v;
  }
  red[threadIdx.x] = s; __syncthreads();
  for (int off = 128; off > 0; off >>= 1) {
    if (threadIdx.x < off) red[threadIdx.x] += red[threadIdx.x + off];
    __syncthreads();
  }
  float mean = red[0] * (1.0f / D_);
  __syncthreads();
  red[threadIdx.x] = s2; __syncthreads();
  for (int off = 128; off > 0; off >>= 1) {
    if (threadIdx.x < off) red[threadIdx.x] += red[threadIdx.x + off];
    __syncthreads();
  }
  float var = red[0] * (1.0f / D_) - mean * mean;
  float inv = rsqrtf(var + 1e-5f);
  for (int d = threadIdx.x; d < D_; d += 256) {
    float v = (xr[d] + yr[d] - mean) * inv * g[d] + bta[d];
    outf[row * D_ + d] = v;
    outb[row * D_ + d] = f2bf(v);
  }
}

// ---------- masked softmax over the QUERY axis (faithful to softmax(dim=1)) ----------
// grid: (S/64, B), block 256. tid&63 -> column k, tid>>6 -> q strider (4-way).
__global__ __launch_bounds__(256) void softmax_q_kernel(const float* __restrict__ att,
                                                        bf16t* __restrict__ attb) {
  int b = blockIdx.y;
  int c = threadIdx.x & 63;
  int k = blockIdx.x * 64 + c;
  int qs = threadIdx.x >> 6;
  const float* base = att + (long)b * S_ * S_;
  bf16t* ob = attb + (long)b * S_ * S_;
  __shared__ float red[4][64];

  float m = -3.4e38f;
  for (int q = qs; q < S_; q += 4)
    if (q >= k) m = fmaxf(m, base[(long)q * S_ + k]);     // causal: valid iff q >= k
  red[qs][c] = m; __syncthreads();
  if (qs == 0) {
    float mm = fmaxf(fmaxf(red[0][c], red[1][c]), fmaxf(red[2][c], red[3][c]));
    red[0][c] = mm;
  }
  __syncthreads();
  m = red[0][c];
  __syncthreads();

  float ssum = 0.f;
  for (int q = qs; q < S_; q += 4)
    if (q >= k) ssum += expf(base[(long)q * S_ + k] - m);
  red[qs][c] = ssum; __syncthreads();
  if (qs == 0) red[0][c] = red[0][c] + red[1][c] + red[2][c] + red[3][c];
  __syncthreads();
  float inv = 1.0f / red[0][c];

  for (int q = qs; q < S_; q += 4) {
    float o = (q >= k) ? expf(base[(long)q * S_ + k] - m) * inv : 0.0f;
    ob[(long)q * S_ + k] = f2bf(o);
  }
}

// ---------- WMMA bf16 GEMM:  C[M,N] = A[M,K] * W[N,K]^T (+bias) ----------
// Both operands contraction-major in memory. 8 waves/block; wave tile 32x64
// (2x4 WMMA tiles); block tile 128x128. All dims used are multiples of 128/32.
template <bool OUT_F32, bool OUT_BF16, bool TRANS_BF16, bool RELU>
__global__ __launch_bounds__(256) void gemm_wmma(const bf16t* __restrict__ A, long lda,
                                                 const bf16t* __restrict__ W, long ldw,
                                                 const float* __restrict__ bias,
                                                 float* __restrict__ Cf, long ldc,
                                                 bf16t* __restrict__ Cb, long ldcb,
                                                 int M, int N, int K) {
  const int tid = threadIdx.x;
  const int lane = tid & 31;
  const int wave = tid >> 5;
  const int wm = wave & 3;                 // 4 waves along M
  const int wn = wave >> 2;                // 2 waves along N
  const int m_base = blockIdx.y * 128 + wm * 32;
  const int n_base = blockIdx.x * 128 + wn * 64;
  const int laneMod = lane & 15;
  const int laneHalf = lane >> 4;          // 0 | 1

  v8f acc[2][4];
#pragma unroll
  for (int i = 0; i < 2; ++i)
#pragma unroll
    for (int j = 0; j < 4; ++j)
#pragma unroll
      for (int e = 0; e < 8; ++e) acc[i][j][e] = 0.0f;

  for (int k0 = 0; k0 < K; k0 += 32) {
    // A tile (16x32 bf16): lane<16 holds K {k0..k0+15}; +16 for upper half
    v16bf a[2];
#pragma unroll
    for (int i = 0; i < 2; ++i) {
      const bf16t* p = A + (long)(m_base + i * 16 + laneMod) * lda + k0 + laneHalf * 16;
      v8bf lo = *(const v8bf*)(p);
      v8bf hi = *(const v8bf*)(p + 8);
#pragma unroll
      for (int e = 0; e < 8; ++e) { a[i][e] = lo[e]; a[i][e + 8] = hi[e]; }
    }
    // B tile (32x16): lane holds column laneMod, K contiguous (+16 for upper half)
#pragma unroll
    for (int j = 0; j < 4; ++j) {
      const bf16t* p = W + (long)(n_base + j * 16 + laneMod) * ldw + k0 + laneHalf * 16;
      v16bf b = *(const v16bf*)p;
#pragma unroll
      for (int i = 0; i < 2; ++i) {
        acc[i][j] = __builtin_amdgcn_wmma_f32_16x16x32_bf16(
            false, a[i], false, b, (short)0, acc[i][j], false, false);
      }
    }
  }

  // C/D 16x16 f32 layout: VGPR r -> row r (+8 for lanes 16..31), col = lane&15
  const int rowOff = laneHalf * 8;
#pragma unroll
  for (int i = 0; i < 2; ++i) {
#pragma unroll
    for (int j = 0; j < 4; ++j) {
      int n = n_base + j * 16 + laneMod;
      float bv = bias ? bias[n] : 0.0f;
#pragma unroll
      for (int r = 0; r < 8; ++r) {
        int m = m_base + i * 16 + rowOff + r;
        float v = acc[i][j][r] + bv;
        if (RELU) v = v > 0.f ? v : 0.f;
        if (OUT_F32) Cf[(long)m * ldc + n] = v;
        if (OUT_BF16) {
          if (TRANS_BF16) Cb[(long)n * ldcb + m] = f2bf(v);
          else            Cb[(long)m * ldcb + n] = f2bf(v);
        }
      }
    }
  }
}

// ---------- host-side attention layer ----------
static void run_attention(const bf16t* Ab,
                          const bf16t* Wq, const float* bq,
                          const bf16t* Wk, const float* bk,
                          const bf16t* Wv, const float* bv,
                          bf16t* Qb, bf16t* Kb, bf16t* VT,
                          float* ATTf, bf16t* ATTb, float* AO,
                          hipStream_t stream) {
  dim3 g1(D_ / 128, (B_ * S_) / 128);
  gemm_wmma<false, true, false, false><<<g1, 256, 0, stream>>>(
      Ab, D_, Wq, D_, bq, nullptr, 0, Qb, D_, B_ * S_, D_, D_);
  gemm_wmma<false, true, false, false><<<g1, 256, 0, stream>>>(
      Ab, D_, Wk, D_, bk, nullptr, 0, Kb, D_, B_ * S_, D_, D_);
  // V written transposed: VT[d][b*S+s]  -> contraction-major operand for att@v
  gemm_wmma<false, true, true, false><<<g1, 256, 0, stream>>>(
      Ab, D_, Wv, D_, bv, nullptr, 0, VT, B_ * S_, B_ * S_, D_, D_);

  dim3 g2(S_ / 128, S_ / 128);
  for (int b = 0; b < B_; ++b)
    gemm_wmma<true, false, false, false><<<g2, 256, 0, stream>>>(
        Qb + (long)b * S_ * D_, D_, Kb + (long)b * S_ * D_, D_, nullptr,
        ATTf + (long)b * S_ * S_, S_, nullptr, 0, S_, S_, D_);

  dim3 gs(S_ / 64, B_);
  softmax_q_kernel<<<gs, 256, 0, stream>>>(ATTf, ATTb);

  dim3 g3(D_ / 128, S_ / 128);
  for (int b = 0; b < B_; ++b)
    gemm_wmma<true, false, false, false><<<g3, 256, 0, stream>>>(
        ATTb + (long)b * S_ * S_, S_, VT + (long)b * S_, B_ * S_, nullptr,
        AO + (long)b * S_ * D_, D_, nullptr, 0, S_, D_, S_);
}

extern "C" void kernel_launch(void* const* d_in, const int* in_sizes, int n_in,
                              void* d_out, int out_size, void* d_ws, size_t ws_size,
                              hipStream_t stream) {
  (void)in_sizes; (void)n_in; (void)out_size; (void)ws_size;

  const int*   x    = (const int*)d_in[0];
  const float* emb  = (const float*)d_in[1];
  const float* pos  = (const float*)d_in[2];
  const float* k1_w = (const float*)d_in[3];  const float* k1_b = (const float*)d_in[4];
  const float* q1_w = (const float*)d_in[5];  const float* q1_b = (const float*)d_in[6];
  const float* v1_w = (const float*)d_in[7];  const float* v1_b = (const float*)d_in[8];
  const float* k2_w = (const float*)d_in[9];  const float* k2_b = (const float*)d_in[10];
  const float* q2_w = (const float*)d_in[11]; const float* q2_b = (const float*)d_in[12];
  const float* v2_w = (const float*)d_in[13]; const float* v2_b = (const float*)d_in[14];
  const float* ln_g = (const float*)d_in[15]; const float* ln_b = (const float*)d_in[16];
  const float* ff_w = (const float*)d_in[17]; const float* ff_b = (const float*)d_in[18];
  const float* ow   = (const float*)d_in[19]; const float* ob   = (const float*)d_in[20];
  float* out = (float*)d_out;

  // ---- workspace arena (150 MiB) ----
  char* ws = (char*)d_ws;
  size_t off = 0;
  auto alloc = [&](size_t bytes) -> char* {
    char* p = ws + off;
    off += (bytes + 255) & ~(size_t)255;
    return p;
  };
  float* A   = (float*)alloc((size_t)B_ * S_ * D_ * 4);   // residual f32 (ping)
  float* A2  = (float*)alloc((size_t)B_ * S_ * D_ * 4);   // residual f32 (pong)
  float* AO  = (float*)alloc((size_t)B_ * S_ * D_ * 4);   // attention f32 out
  bf16t* Ab  = (bf16t*)alloc((size_t)B_ * S_ * D_ * 2);   // current activation bf16
  bf16t* Qb  = (bf16t*)alloc((size_t)B_ * S_ * D_ * 2);   // q bf16 / ff2 bf16
  bf16t* T1b = (bf16t*)alloc((size_t)B_ * S_ * D_ * 2);   // relu(ff1) bf16
  bf16t* Wq1 = (bf16t*)alloc((size_t)D_ * D_ * 2);
  bf16t* Wk1 = (bf16t*)alloc((size_t)D_ * D_ * 2);
  bf16t* Wv1 = (bf16t*)alloc((size_t)D_ * D_ * 2);
  bf16t* Wq2 = (bf16t*)alloc((size_t)D_ * D_ * 2);
  bf16t* Wk2 = (bf16t*)alloc((size_t)D_ * D_ * 2);
  bf16t* Wv2 = (bf16t*)alloc((size_t)D_ * D_ * 2);
  bf16t* Wff = (bf16t*)alloc((size_t)D_ * D_ * 2);
  // region reused: [Kb | VT | ATTf | ATTb] during attention, then OWb for vocab proj
  char* region = alloc(67108864);
  bf16t* Kb   = (bf16t*)region;                            // 8 MiB
  bf16t* VT   = (bf16t*)(region + 8388608);                // 8 MiB
  float* ATTf = (float*)(region + 16777216);               // 32 MiB
  bf16t* ATTb = (bf16t*)(region + 50331648);               // 16 MiB
  bf16t* OWb  = (bf16t*)region;                            // 62.5 MiB (after attn)

  // ---- convert D x D weights to bf16 ----
  const long wn = (long)D_ * D_;
  const int cgrid = (int)((wn + 255) / 256);
  cvt_bf16_kernel<<<cgrid, 256, 0, stream>>>(q1_w, Wq1, wn);
  cvt_bf16_kernel<<<cgrid, 256, 0, stream>>>(k1_w, Wk1, wn);
  cvt_bf16_kernel<<<cgrid, 256, 0, stream>>>(v1_w, Wv1, wn);
  cvt_bf16_kernel<<<cgrid, 256, 0, stream>>>(q2_w, Wq2, wn);
  cvt_bf16_kernel<<<cgrid, 256, 0, stream>>>(k2_w, Wk2, wn);
  cvt_bf16_kernel<<<cgrid, 256, 0, stream>>>(v2_w, Wv2, wn);
  cvt_bf16_kernel<<<cgrid, 256, 0, stream>>>(ff_w, Wff, wn);

  // ---- h0 = emb[x] + pos ----
  const long tot = (long)B_ * S_ * D_;
  embed_kernel<<<(int)((tot + 255) / 256), 256, 0, stream>>>(x, emb, pos, A, Ab, tot);

  // ---- layer 1 ----
  run_attention(Ab, Wq1, q1_b, Wk1, k1_b, Wv1, v1_b, Qb, Kb, VT, ATTf, ATTb, AO, stream);
  add_ln_kernel<<<B_ * S_, 256, 0, stream>>>(AO, A, ln_g, ln_b, A2, Ab);   // h1 -> A2/Ab

  // ---- layer 2 ----
  run_attention(Ab, Wq2, q2_b, Wk2, k2_b, Wv2, v2_b, Qb, Kb, VT, ATTf, ATTb, AO, stream);
  add_ln_kernel<<<B_ * S_, 256, 0, stream>>>(AO, A2, ln_g, ln_b, A, Ab);   // h2 -> A/Ab

  // ---- feed-forward: same linear twice, ReLU between ----
  dim3 gff(D_ / 128, (B_ * S_) / 128);
  gemm_wmma<false, true, false, true><<<gff, 256, 0, stream>>>(
      Ab, D_, Wff, D_, ff_b, nullptr, 0, T1b, D_, B_ * S_, D_, D_);
  gemm_wmma<false, true, false, false><<<gff, 256, 0, stream>>>(
      T1b, D_, Wff, D_, ff_b, nullptr, 0, Qb, D_, B_ * S_, D_, D_);

  // ---- vocab projection (dominant GEMM, 268 GFLOP) ----
  const long vwn = (long)V_ * D_;
  cvt_bf16_kernel<<<(int)((vwn + 255) / 256), 256, 0, stream>>>(ow, OWb, vwn);
  dim3 gout(V_ / 128, (B_ * S_) / 128);
  gemm_wmma<true, false, false, false><<<gout, 256, 0, stream>>>(
      Qb, D_, OWb, D_, ob, out, V_, nullptr, 0, B_ * S_, V_, D_);
}
